// TransformerDecoderUnit_89764816486653
// MI455X (gfx1250) — matrix operand order repaired and checked by reading, not compiled, not run on hardware
//
#include <hip/hip_runtime.h>
#include <hip/hip_bf16.h>
#include <math.h>

typedef __attribute__((ext_vector_type(16))) __bf16 v16bf;
typedef __attribute__((ext_vector_type(8)))  float  v8f;
typedef __attribute__((ext_vector_type(4)))  int    i4v;

static constexpr int kB  = 2;
static constexpr int kC  = 256;
static constexpr int kHW = 2304;   // 48*48
static constexpr int kNH = 8;
static constexpr int kDH = 32;

#if defined(__has_builtin)
#if __has_builtin(__builtin_amdgcn_global_load_async_to_lds_b128)
#define HAVE_ASYNC_LDS 1
#endif
#endif

#ifdef HAVE_ASYNC_LDS
// Probe-confirmed signature: (int4 AS(1)* src, int4 AS(3)* dst, imm offset, imm cpol)
typedef __attribute__((address_space(1))) i4v gas_i4;
typedef __attribute__((address_space(3))) i4v las_i4;
__device__ __forceinline__ void wait_async0() {
#if __has_builtin(__builtin_amdgcn_s_wait_asynccnt)
  __builtin_amdgcn_s_wait_asynccnt(0);
#else
  asm volatile("s_wait_asynccnt 0x0" ::: "memory");
#endif
}
#endif

__device__ __forceinline__ __bf16 f2bf(float f) {
  union { float f; unsigned u; } in; in.f = f;
  unsigned r = (in.u + 0x7FFFu + ((in.u >> 16) & 1u)) >> 16;
  union { unsigned short s; __bf16 b; } out; out.s = (unsigned short)r;
  return out.b;
}

__device__ __forceinline__ v8f v8f_zero() {
  v8f z = {0.f,0.f,0.f,0.f,0.f,0.f,0.f,0.f};
  return z;
}

// Pack two 8-element (16B) chunks into the 16-half WMMA operand register file.
__device__ __forceinline__ v16bf ldbf16x16(const __bf16* p0, const __bf16* p1) {
  union { v16bf v; float4 f[2]; } u;
  u.f[0] = *(const float4*)p0;
  u.f[1] = *(const float4*)p1;
  return u.v;
}

// ---------------------------------------------------------------------------
// Weight fp32 -> bf16 (6 matrices of 256x256, packed consecutively)
// ---------------------------------------------------------------------------
__global__ void k_convert_weights(const float* __restrict__ w0, const float* __restrict__ w1,
                                  const float* __restrict__ w2, const float* __restrict__ w3,
                                  const float* __restrict__ w4, const float* __restrict__ w5,
                                  __bf16* __restrict__ dst) {
  int i = blockIdx.x * blockDim.x + threadIdx.x;
  if (i >= 6 * kC * kC) return;
  int m = i >> 16, j = i & 65535;
  const float* src = (m == 0) ? w0 : (m == 1) ? w1 : (m == 2) ? w2
                   : (m == 3) ? w3 : (m == 4) ? w4 : w5;
  dst[i] = f2bf(src[j]);
}

// ---------------------------------------------------------------------------
// qp = q + pos, kp = k + pos, v -> bf16, all TOKEN-MAJOR (b, hw, c)
// ---------------------------------------------------------------------------
__global__ void k_prep(const float* __restrict__ q, const float* __restrict__ k,
                       const float* __restrict__ v,
                       __bf16* __restrict__ qp, __bf16* __restrict__ kp,
                       __bf16* __restrict__ vb) {
  int i = blockIdx.x * blockDim.x + threadIdx.x;     // (bb, t, c) token-major index
  if (i >= kB * kC * kHW) return;
  int c  = i & (kC - 1);
  int t  = (i >> 8) % kHW;
  int bb = i / (kC * kHW);
  int y = t / 48, x = t % 48;
  int pi = (c < 128) ? c : (c - 128);
  float norm  = 6.28318530717958647f / (48.0f + 1e-6f);
  float coord = (c < 128) ? ((float)(y + 1) * norm) : ((float)(x + 1) * norm);
  float dimt = __powf(10000.0f, (float)(2 * (pi >> 1)) * (1.0f / 128.0f));
  float p = coord / dimt;
  float pos = (pi & 1) ? __cosf(p) : __sinf(p);
  size_t src = ((size_t)bb * kC + c) * kHW + t;      // inputs are (b, c, hw)
  qp[i] = f2bf(q[src] + pos);
  kp[i] = f2bf(k[src] + pos);
  vb[i] = f2bf(v[src]);
}

// ---------------------------------------------------------------------------
// WMMA GEMM, token-major: out[t][o] = sum_c X_tm[t][c] * W[o][c]
//   M = tokens (A-frag: channel run of one token, contiguous)
//   N = out-channels (B-frag: one row of row-major W, contiguous)
// Block: 128 tokens x 128 channels; wave tile 32x64 (2x4 WMMA tiles).
// A tile staged via async global->LDS DMA when available.
//   mode 0: bf16 per-head token-major [(b,nh,hw,dh)], scaled by oscale
//   mode 1: bf16 token-major          [(b,hw,c)]
//   mode 2: f32 + bf16 token-major
//   mode 3: +bias, relu, bf16 token-major
//   mode 4: +bias +residual, f32 token-major
// ---------------------------------------------------------------------------
static constexpr int GT = 128, GN = 128, GK = 32;

__global__ __launch_bounds__(256)
void k_gemm(const __bf16* __restrict__ W, const __bf16* __restrict__ X,
            int mode, const float* __restrict__ bias,
            const float* __restrict__ resid,
            float* __restrict__ outF, __bf16* __restrict__ outB, float oscale) {
  __shared__ __align__(16) __bf16 As[GT][GK];   // 8 KB

  const int tid  = threadIdx.x;
  const int lane = tid & 31;
  const int wave = tid >> 5;
  const int hi   = lane >> 4;
  const int ln   = lane & 15;
  const int k0   = 8 * hi;
  const int bb   = blockIdx.z;
  const int tb   = blockIdx.x * GT;
  const int nb   = blockIdx.y * GN;
  const int wm   = wave & 3;    // 4 waves along tokens   -> 128
  const int wn   = wave >> 2;   // 2 waves along channels -> 128
  const __bf16* Xb = X + (size_t)bb * kHW * kC;

  v8f acc[2][4];
  #pragma unroll
  for (int a = 0; a < 2; a++)
    #pragma unroll
    for (int b = 0; b < 4; b++) acc[a][b] = v8f_zero();

  const int arow = tid >> 1;          // 0..127
  const int acol = (tid & 1) * 16;    // 0 or 16

  for (int kk = 0; kk < kC; kk += GK) {
    // A tile: 128 tokens x 32 K, each thread copies 32 bytes
    const __bf16* src = &Xb[(size_t)(tb + arow) * kC + kk + acol];
#ifdef HAVE_ASYNC_LDS
    __builtin_amdgcn_global_load_async_to_lds_b128(
        (gas_i4*)src, (las_i4*)&As[arow][acol], 0, 0);
    __builtin_amdgcn_global_load_async_to_lds_b128(
        (gas_i4*)(src + 8), (las_i4*)&As[arow][acol + 8], 0, 0);
    wait_async0();
#else
    *(float4*)&As[arow][acol]     = *(const float4*)src;
    *(float4*)&As[arow][acol + 8] = *(const float4*)(src + 8);
#endif
    __syncthreads();

    v16bf af[2];
    #pragma unroll
    for (int mi = 0; mi < 2; mi++) {
      int row = wm * 32 + mi * 16 + ln;
      af[mi] = ldbf16x16(&As[row][k0], &As[row][16 + k0]);
    }
    #pragma unroll
    for (int ni = 0; ni < 4; ni++) {
      int o = nb + wn * 64 + ni * 16 + ln;
      v16bf bfr = ldbf16x16(&W[(size_t)o * kC + kk + k0],
                            &W[(size_t)o * kC + kk + 16 + k0]);
      #pragma unroll
      for (int mi = 0; mi < 2; mi++)
        acc[mi][ni] = __builtin_amdgcn_wmma_f32_16x16x32_bf16(
            false, af[mi], false, bfr, (short)0, acc[mi][ni], false, false);
    }
    __syncthreads();
  }

  // epilogue: C layout -> row (token) = r + 8*hi, col (channel) = ln
  #pragma unroll
  for (int mi = 0; mi < 2; mi++) {
    #pragma unroll
    for (int ni = 0; ni < 4; ni++) {
      #pragma unroll
      for (int r = 0; r < 8; r++) {
        int t = tb + wm * 32 + mi * 16 + r + 8 * hi;
        int o = nb + wn * 64 + ni * 16 + ln;
        float vv = acc[mi][ni][r] * oscale;
        if (mode == 0) {
          int head = o >> 5, dim = o & 31;
          outB[(((size_t)bb * kNH + head) * kHW + t) * kDH + dim] = f2bf(vv);
        } else if (mode == 1) {
          outB[((size_t)bb * kHW + t) * kC + o] = f2bf(vv);
        } else if (mode == 2) {
          size_t idx = ((size_t)bb * kHW + t) * kC + o;
          outF[idx] = vv;
          outB[idx] = f2bf(vv);
        } else if (mode == 3) {
          float h = vv + bias[o];
          h = h > 0.f ? h : 0.f;
          outB[((size_t)bb * kHW + t) * kC + o] = f2bf(h);
        } else {
          size_t idx = ((size_t)bb * kHW + t) * kC + o;
          outF[idx] = vv + bias[o] + resid[idx];
        }
      }
    }
  }
}

// ---------------------------------------------------------------------------
// Flash attention.
//  Q,K: (b,nh,hw,dh) bf16, Q pre-scaled by 1/sqrt(d).
//  V:   token-major (b,hw,c) bf16 -> 32x32 chunk transposed through LDS so the
//       PV B-fragment (two 8-key runs per dim column) is contiguous.
//  O:   token-major (b,hw,c) bf16.
// Each wave owns 16 queries; 32-key chunks with online softmax.
// ---------------------------------------------------------------------------
__global__ __launch_bounds__(256)
void k_flash(const __bf16* __restrict__ Q, const __bf16* __restrict__ K,
             const __bf16* __restrict__ Vtm, __bf16* __restrict__ Otm) {
  __shared__ __align__(16) __bf16 Pld[8][16 * 32];  // per-wave P slab
  __shared__ __align__(16) __bf16 VT[32][40];       // [dim][key], padded stride

  const int tid   = threadIdx.x;
  const int lane  = tid & 31;
  const int wave  = tid >> 5;
  const int hi    = lane >> 4;
  const int ln    = lane & 15;
  const int k0    = 8 * hi;
  const int head  = blockIdx.y;
  const int bb    = blockIdx.z;
  const int qbase = blockIdx.x * 128 + wave * 16;

  const __bf16* Qh = Q + (((size_t)bb * kNH + head) * kHW) * kDH;
  const __bf16* Kh = K + (((size_t)bb * kNH + head) * kHW) * kDH;

  v16bf qf = ldbf16x16(&Qh[(size_t)(qbase + ln) * kDH + k0],
                       &Qh[(size_t)(qbase + ln) * kDH + 16 + k0]);

  float mrun[8], lrun[8];
  v8f acc0 = v8f_zero(), acc1 = v8f_zero();
  #pragma unroll
  for (int r = 0; r < 8; r++) { mrun[r] = -1e30f; lrun[r] = 0.f; }

  __bf16* pslab = &Pld[wave][0];
  const int vkey = tid >> 3;          // 0..31
  const int vd0  = (tid & 7) * 4;     // 0..28

  for (int kc = 0; kc < kHW; kc += 32) {
    // fetch this block's V chunk (32 keys x 32 dims, token-major rows)
    union { float2 f; __bf16 h[4]; } uv;
    uv.f = *(const float2*)&Vtm[((size_t)bb * kHW + kc + vkey) * kC + head * kDH + vd0];
    __syncthreads();                  // previous chunk's VT reads complete
    #pragma unroll
    for (int j = 0; j < 4; j++) VT[vd0 + j][vkey] = uv.h[j];

    // scores: K B-fragments (lane -> key column, dims k0..k0+7 and 16+k0..23+k0)
    v16bf kf0 = ldbf16x16(&Kh[(size_t)(kc + ln) * kDH + k0],
                          &Kh[(size_t)(kc + ln) * kDH + 16 + k0]);
    v16bf kf1 = ldbf16x16(&Kh[(size_t)(kc + 16 + ln) * kDH + k0],
                          &Kh[(size_t)(kc + 16 + ln) * kDH + 16 + k0]);
    v8f s0 = __builtin_amdgcn_wmma_f32_16x16x32_bf16(false, qf, false, kf0,
                                                     (short)0, v8f_zero(), false, false);
    v8f s1 = __builtin_amdgcn_wmma_f32_16x16x32_bf16(false, qf, false, kf1,
                                                     (short)0, v8f_zero(), false, false);
    #pragma unroll
    for (int r = 0; r < 8; r++) {
      float mv = fmaxf(s0[r], s1[r]);
      #pragma unroll
      for (int off = 1; off < 16; off <<= 1) mv = fmaxf(mv, __shfl_xor(mv, off, 32));
      float mnew   = fmaxf(mrun[r], mv);
      float cscale = __expf(mrun[r] - mnew);
      mrun[r] = mnew;
      float p0 = __expf(s0[r] - mnew);
      float p1 = __expf(s1[r] - mnew);
      float rs = p0 + p1;
      #pragma unroll
      for (int off = 1; off < 16; off <<= 1) rs += __shfl_xor(rs, off, 32);
      lrun[r] = lrun[r] * cscale + rs;
      acc0[r] *= cscale;
      acc1[r] *= cscale;
      int ml = r + 8 * hi;            // stage P (C layout) row-major 16x32
      pslab[ml * 32 + ln]      = f2bf(p0);
      pslab[ml * 32 + 16 + ln] = f2bf(p1);
    }
    // P as A-fragment (same-wave DS ops are in order)
    v16bf pf = ldbf16x16(&pslab[ln * 32 + k0], &pslab[ln * 32 + 16 + k0]);

    __syncthreads();                  // VT fully written
    v16bf vf0 = ldbf16x16(&VT[ln][k0],      &VT[ln][16 + k0]);
    v16bf vf1 = ldbf16x16(&VT[16 + ln][k0], &VT[16 + ln][16 + k0]);
    acc0 = __builtin_amdgcn_wmma_f32_16x16x32_bf16(false, pf, false, vf0,
                                                   (short)0, acc0, false, false);
    acc1 = __builtin_amdgcn_wmma_f32_16x16x32_bf16(false, pf, false, vf1,
                                                   (short)0, acc1, false, false);
  }

  __bf16* Ob = Otm + (size_t)bb * kHW * kC + (size_t)head * kDH;
  #pragma unroll
  for (int r = 0; r < 8; r++) {
    int qrow = qbase + r + 8 * hi;
    float inv = 1.0f / lrun[r];
    Ob[(size_t)qrow * kC + ln]      = f2bf(acc0[r] * inv);
    Ob[(size_t)qrow * kC + 16 + ln] = f2bf(acc1[r] * inv);
  }
}

// ---------------------------------------------------------------------------
// BatchNorm over token-major y: per-channel partial sums (coalesced: thread =
// channel), accumulated with f32 global atomics; finalized in apply kernel.
// ---------------------------------------------------------------------------
__global__ void k_zero(float* __restrict__ p, int n) {
  int i = blockIdx.x * blockDim.x + threadIdx.x;
  if (i < n) p[i] = 0.f;
}

__global__ __launch_bounds__(256)
void k_bn_partial(const float* __restrict__ y, float* __restrict__ sum,
                  float* __restrict__ sq) {
  int ch   = threadIdx.x;
  int row0 = blockIdx.x * 128;      // flat (b,t) rows, 4608 total
  float s = 0.f, q = 0.f;
  for (int r = 0; r < 128; r++) {
    float v = y[(size_t)(row0 + r) * kC + ch];
    s += v; q += v * v;
  }
  atomicAdd(&sum[ch], s);
  atomicAdd(&sq[ch], q);
}

__global__ void k_bn_apply(const float* __restrict__ y, const float* __restrict__ sum,
                           const float* __restrict__ sq, const float* __restrict__ gamma,
                           const float* __restrict__ beta, float* __restrict__ out) {
  int j = blockIdx.x * blockDim.x + threadIdx.x;     // output index, (b,c,hw)
  if (j >= kB * kC * kHW) return;
  int t  = j % kHW;
  int ch = (j / kHW) % kC;
  int bb = j / (kC * kHW);
  const float invN = 1.0f / (float)(kB * kHW);
  float m   = sum[ch] * invN;
  float var = sq[ch] * invN - m * m;
  float v   = y[((size_t)bb * kHW + t) * kC + ch];
  out[j] = (v - m) * rsqrtf(var + 1e-5f) * gamma[ch] + beta[ch];
}

// ---------------------------------------------------------------------------
extern "C" void kernel_launch(void* const* d_in, const int* in_sizes, int n_in,
                              void* d_out, int out_size, void* d_ws, size_t ws_size,
                              hipStream_t stream) {
  (void)in_sizes; (void)n_in; (void)out_size; (void)ws_size;
  const float* q     = (const float*)d_in[0];
  const float* k     = (const float*)d_in[1];
  const float* v     = (const float*)d_in[2];
  const float* w_q   = (const float*)d_in[3];
  const float* w_k   = (const float*)d_in[4];
  const float* w_v   = (const float*)d_in[5];
  const float* w_fc  = (const float*)d_in[6];
  const float* w1    = (const float*)d_in[7];
  const float* b1    = (const float*)d_in[8];
  const float* w2    = (const float*)d_in[9];
  const float* b2    = (const float*)d_in[10];
  const float* gamma = (const float*)d_in[11];
  const float* beta  = (const float*)d_in[12];

  const size_t actBF  = (size_t)kB * kC * kHW * sizeof(__bf16);
  const size_t actF32 = (size_t)kB * kC * kHW * sizeof(float);

  char* ws = (char*)d_ws;
  size_t off = 0;
  auto alloc = [&](size_t bytes) -> char* {
    char* p = ws + off;
    off += (bytes + 255) & ~(size_t)255;
    return p;
  };
  __bf16* wbf    = (__bf16*)alloc(6 * kC * kC * sizeof(__bf16));
  __bf16* qp     = (__bf16*)alloc(actBF);   // token-major
  __bf16* kp     = (__bf16*)alloc(actBF);
  __bf16* vbf    = (__bf16*)alloc(actBF);
  __bf16* Qtok   = (__bf16*)alloc(actBF);   // (b,nh,hw,dh)
  __bf16* Ktok   = (__bf16*)alloc(actBF);
  __bf16* Vtm    = (__bf16*)alloc(actBF);   // token-major
  __bf16* Obf    = (__bf16*)alloc(actBF);   // token-major
  __bf16* fc_bf  = (__bf16*)alloc(actBF);   // token-major
  __bf16* hid_bf = (__bf16*)alloc(actBF);   // token-major
  float*  fc_f32 = (float*)alloc(actF32);   // token-major
  float*  y_f32  = (float*)alloc(actF32);   // token-major
  float*  sump   = (float*)alloc(kC * sizeof(float));
  float*  sqp    = (float*)alloc(kC * sizeof(float));

  __bf16* wq_bf  = wbf + 0 * kC * kC;
  __bf16* wk_bf  = wbf + 1 * kC * kC;
  __bf16* wv_bf  = wbf + 2 * kC * kC;
  __bf16* wfc_bf = wbf + 3 * kC * kC;
  __bf16* w1_bf  = wbf + 4 * kC * kC;
  __bf16* w2_bf  = wbf + 5 * kC * kC;

  k_zero<<<2, 256, 0, stream>>>(sump, 2 * kC);   // sump+sqp contiguous

  k_convert_weights<<<(6 * kC * kC + 255) / 256, 256, 0, stream>>>(
      w_q, w_k, w_v, w_fc, w1, w2, wbf);

  k_prep<<<(kB * kC * kHW + 255) / 256, 256, 0, stream>>>(q, k, v, qp, kp, vbf);

  dim3 gGemm(kHW / GT, kC / GN, kB);             // (18, 2, 2)
  const float qscale = 0.17677669529663687f;     // 1/sqrt(32)

  k_gemm<<<gGemm, 256, 0, stream>>>(wq_bf, qp,  0, nullptr, nullptr, nullptr, Qtok, qscale);
  k_gemm<<<gGemm, 256, 0, stream>>>(wk_bf, kp,  0, nullptr, nullptr, nullptr, Ktok, 1.0f);
  k_gemm<<<gGemm, 256, 0, stream>>>(wv_bf, vbf, 1, nullptr, nullptr, nullptr, Vtm,  1.0f);

  dim3 gAttn(kHW / 128, kNH, kB);
  k_flash<<<gAttn, 256, 0, stream>>>(Qtok, Ktok, Vtm, Obf);

  k_gemm<<<gGemm, 256, 0, stream>>>(wfc_bf, Obf,   2, nullptr, nullptr, fc_f32, fc_bf, 1.0f);
  k_gemm<<<gGemm, 256, 0, stream>>>(w1_bf,  fc_bf, 3, b1, nullptr, nullptr, hid_bf, 1.0f);
  k_gemm<<<gGemm, 256, 0, stream>>>(w2_bf,  hid_bf,4, b2, fc_f32, y_f32, nullptr, 1.0f);

  k_bn_partial<<<(kB * kHW) / 128, 256, 0, stream>>>(y_f32, sump, sqp);
  k_bn_apply<<<(kB * kC * kHW + 255) / 256, 256, 0, stream>>>(
      y_f32, sump, sqp, gamma, beta, (float*)d_out);
}
